// LatentSDE_89352499626553
// MI455X (gfx1250) — matrix operand order repaired and checked by reading, not compiled, not run on hardware
//
#include <hip/hip_runtime.h>
#include <math.h>

typedef __attribute__((ext_vector_type(16))) _Float16 v16h;
typedef __attribute__((ext_vector_type(8)))  _Float16 v8h;
typedef __attribute__((ext_vector_type(8)))  float    v8f;

#define WMMA_F16(a, b, c) \
    __builtin_amdgcn_wmma_f32_16x16x32_f16(false, (a), false, (b), (short)0, (c), false, false)

// Branch-free tanh: prefer the CDNA5 hardware V_TANH_F32 (TRANS32 op), else a
// 6-op branchless sequence (exp2 + rcp + copysign). Never the branchy libm path.
static __device__ __forceinline__ float fast_tanhf(float x) {
#if __has_builtin(__builtin_amdgcn_tanhf)
    return __builtin_amdgcn_tanhf(x);
#elif __has_builtin(__builtin_amdgcn_tanh_f32)
    return __builtin_amdgcn_tanh_f32(x);
#else
    // tanh(|x|) = (1 - e) / (1 + e),  e = exp(-2|x|) = 2^(-2|x|*log2(e))
    const float ax = __builtin_fabsf(x);
    const float e  = __builtin_amdgcn_exp2f(ax * -2.8853900818f);
    const float t  = (1.0f - e) * __builtin_amdgcn_rcpf(1.0f + e);
    return __builtin_copysignf(t, x);
#endif
}

// LatentSDE Euler-Maruyama: z_{i+1} = z_i + MLP([t_i, z_i])*dt + std*dW_i*sqrt(dt)
// One 16-row batch tile per workgroup; 4 waves split the N dimension of the MLP.
// Weights live in LDS as f16; per-wave B fragments live in registers across all T steps.
__global__ __launch_bounds__(128)
void latent_sde_wmma(const float* __restrict__ z0, const float* __restrict__ ts,
                     const float* __restrict__ dWg, const float* __restrict__ W1,
                     const float* __restrict__ b1,  const float* __restrict__ W2,
                     const float* __restrict__ b2,  const float* __restrict__ W3,
                     const float* __restrict__ b3,  const float* __restrict__ logstd,
                     float* __restrict__ out)
{
    constexpr int B = 8192, T = 256;

    __shared__ _Float16 sW1[32 * 64];            // W1 rows 1..32 (z part), [k][h], f16
    __shared__ _Float16 sW2[64 * 64];            // [k][h]
    __shared__ _Float16 sW3[64 * 32];            // [k][d]
    __shared__ float sW1t[64];                   // W1 row 0 (the t input row)
    __shared__ float sb1[64], sb2[64], sb3[32], sstd[32];
    __shared__ float sz[16 * 32];                // f32 state tile, row-major
    __shared__ __align__(16) _Float16 sh1[16 * 64]; // h1 staging
    __shared__ __align__(16) _Float16 sh2[16 * 64]; // h2 staging (separate buffer -> fewer barriers)
    __shared__ __align__(16) float sdW[2][16 * 32]; // double-buffered noise tile

    const int tid  = threadIdx.x;
    const int wave = tid >> 5;
    const int lane = tid & 31;
    const int hi   = lane >> 4;      // lane group: 0 => lanes 0-15, 1 => lanes 16-31
    const int l15  = lane & 15;
    const int row0 = blockIdx.x * 16;            // global batch row base of this tile

    // ---------------- prologue: stage weights (f32 -> f16) ----------------
    for (int i = tid; i < 32 * 64; i += 128) {
        int k = i >> 6, h = i & 63;
        sW1[i] = (_Float16)W1[(k + 1) * 64 + h];
    }
    for (int i = tid; i < 64 * 64; i += 128) sW2[i] = (_Float16)W2[i];
    for (int i = tid; i < 64 * 32; i += 128) sW3[i] = (_Float16)W3[i];
    if (tid < 64) { sW1t[tid] = W1[tid]; sb1[tid] = b1[tid]; sb2[tid] = b2[tid]; }
    if (tid < 32) { sb3[tid] = b3[tid]; sstd[tid] = expf(logstd[tid]); }

    // z0 tile -> sz, and emit output slice 0 (= z0)
    for (int i = tid; i < 512; i += 128) {
        float v = z0[row0 * 32 + i];
        sz[i] = v;
        out[row0 * 32 + i] = v;                  // ((0*B + row0)*32 + i)
    }
    // preload dW[0] tile (rows of the tile are contiguous in (T-1,B,D))
    for (int i = tid; i < 512; i += 128)
        sdW[0][i] = dWg[(size_t)row0 * 32 + i];
    __syncthreads();

    // ---------------- per-wave resident B fragments ----------------
    // B 32x16 f16 fragment: half i <-> K = 8*hi + (i<8 ? i : i+8), column N = l15 (+16*ntile).
    const int klo  = hi * 8;
    const int colW = wave * 16 + l15;            // this wave's output column (layers 1/2)
    v16h B1f, B2f0, B2f1, B3f0 = {}, B3f1 = {};
    #pragma unroll
    for (int i = 0; i < 16; ++i) B1f[i]  = sW1[(klo + i + (i & 8)) * 64 + colW];
    #pragma unroll
    for (int i = 0; i < 16; ++i) B2f0[i] = sW2[(klo + i + (i & 8)) * 64 + colW];
    #pragma unroll
    for (int i = 0; i < 16; ++i) B2f1[i] = sW2[(32 + klo + i + (i & 8)) * 64 + colW];
    if (wave < 2) {
        #pragma unroll
        for (int i = 0; i < 16; ++i) B3f0[i] = sW3[(klo + i + (i & 8)) * 32 + colW];
        #pragma unroll
        for (int i = 0; i < 16; ++i) B3f1[i] = sW3[(32 + klo + i + (i & 8)) * 32 + colW];
    }
    const float stdc = (wave < 2) ? sstd[colW] : 0.0f;

    // C/D fragment geometry: element v lives at row (hi*8 + v), col colW.
    const int shbase = (hi * 8) * 64 + colW;     // store base into sh* (stride 64 per row)

    // ---------------- time loop ----------------
    for (int step = 0; step < T - 1; ++step) {
        const float t    = ts[step];             // uniform -> scalar loads
        const float dt   = ts[step + 1] - t;
        const float sqdt = sqrtf(dt);

        // A fragment of z (16x32 f16): half i <-> K = 8*hi + (i<8 ? i : i+8), row l15
        v16h Az;
        {
            const int rb = l15 * 32 + hi * 8;
            #pragma unroll
            for (int i = 0; i < 8; ++i) Az[i]     = (_Float16)sz[rb + i];
            #pragma unroll
            for (int i = 0; i < 8; ++i) Az[i + 8] = (_Float16)sz[rb + 16 + i];
        }

        // ---- layer 1: h1 = tanh(z @ W1[1:] + (b1 + t*W1[0])) ----
        const float bias1 = sb1[colW] + t * sW1t[colW];
        v8f c;
        #pragma unroll
        for (int v = 0; v < 8; ++v) c[v] = bias1;
        c = WMMA_F16(Az, B1f, c);
        #pragma unroll
        for (int v = 0; v < 8; ++v) sh1[shbase + v * 64] = (_Float16)fast_tanhf(c[v]);
        __syncthreads();                         // (1) h1 visible

        // gather h1 as two A fragments (K = 0..31 and 32..63); contiguous 16B runs
        v16h A2a, A2b;
        {
            const int rb = l15 * 64 + hi * 8;
            v8h x0 = *(const v8h*)(sh1 + rb);
            v8h x1 = *(const v8h*)(sh1 + rb + 16);
            v8h x2 = *(const v8h*)(sh1 + rb + 32);
            v8h x3 = *(const v8h*)(sh1 + rb + 48);
            #pragma unroll
            for (int i = 0; i < 8; ++i) { A2a[i] = x0[i]; A2a[i + 8] = x1[i];
                                          A2b[i] = x2[i]; A2b[i + 8] = x3[i]; }
        }

        // ---- layer 2: h2 = tanh(h1 @ W2 + b2) -> separate buffer, no barrier needed ----
        const float bias2 = sb2[colW];
        #pragma unroll
        for (int v = 0; v < 8; ++v) c[v] = bias2;
        c = WMMA_F16(A2a, B2f0, c);
        c = WMMA_F16(A2b, B2f1, c);
        #pragma unroll
        for (int v = 0; v < 8; ++v) sh2[shbase + v * 64] = (_Float16)fast_tanhf(c[v]);
        __syncthreads();                         // (2) h2 visible (and all sh1 reads done)

        if (wave < 2) {
            // ---- layer 3 (N=32, waves 0-1): f = h2 @ W3 + b3 ----
            v16h A3a, A3b;
            {
                const int rb = l15 * 64 + hi * 8;
                v8h x0 = *(const v8h*)(sh2 + rb);
                v8h x1 = *(const v8h*)(sh2 + rb + 16);
                v8h x2 = *(const v8h*)(sh2 + rb + 32);
                v8h x3 = *(const v8h*)(sh2 + rb + 48);
                #pragma unroll
                for (int i = 0; i < 8; ++i) { A3a[i] = x0[i]; A3a[i + 8] = x1[i];
                                              A3b[i] = x2[i]; A3b[i + 8] = x3[i]; }
            }
            const float bias3 = sb3[colW];
            #pragma unroll
            for (int v = 0; v < 8; ++v) c[v] = bias3;
            c = WMMA_F16(A3a, B3f0, c);
            c = WMMA_F16(A3b, B3f1, c);

            // ---- Euler-Maruyama update (f32 state) + output store ----
            const int cur = step & 1;
            const size_t obase = ((size_t)(step + 1) * B + row0) * 32;
            #pragma unroll
            for (int v = 0; v < 8; ++v) {
                const int idx = (hi * 8 + v) * 32 + colW;
                const float zn = sz[idx] + c[v] * dt + stdc * sdW[cur][idx] * sqdt;
                sz[idx] = zn;
                out[obase + idx] = zn;
            }
        } else {
            // waves 2-3: double-buffer next noise tile into LDS
            if (step + 1 < T - 1) {
                const int p = tid - 64;          // 0..63, 2 float4 each = 512 floats
                const float4* s4 = (const float4*)(dWg + ((size_t)(step + 1) * B + row0) * 32);
                float4* d4 = (float4*)sdW[(step + 1) & 1];
                d4[2 * p]     = s4[2 * p];
                d4[2 * p + 1] = s4[2 * p + 1];
            }
            if (step + 2 < T - 1)                // warm GL2 for the tile after that
                __builtin_prefetch(dWg + ((size_t)(step + 2) * B + row0) * 32 + (tid - 64) * 8, 0, 1);
        }
        __syncthreads();                         // (3) sz/sdW/sh2 ready for next step
    }
}

extern "C" void kernel_launch(void* const* d_in, const int* in_sizes, int n_in,
                              void* d_out, int out_size, void* d_ws, size_t ws_size,
                              hipStream_t stream) {
    (void)in_sizes; (void)n_in; (void)d_ws; (void)ws_size; (void)out_size;
    const float* z0 = (const float*)d_in[0];
    const float* ts = (const float*)d_in[1];
    const float* dW = (const float*)d_in[2];
    const float* W1 = (const float*)d_in[3];
    const float* b1 = (const float*)d_in[4];
    const float* W2 = (const float*)d_in[5];
    const float* b2 = (const float*)d_in[6];
    const float* W3 = (const float*)d_in[7];
    const float* b3 = (const float*)d_in[8];
    const float* ls = (const float*)d_in[9];
    float* out = (float*)d_out;
    // 8192 rows / 16-row tile = 512 workgroups of 128 threads (4 waves)
    latent_sde_wmma<<<512, 128, 0, stream>>>(z0, ts, dW, W1, b1, W2, b2, W3, b3, ls, out);
}
